// MultiHeadAttention_70841190580799
// MI455X (gfx1250) — matrix operand order repaired and checked by reading, not compiled, not run on hardware
//
#include <hip/hip_runtime.h>

// ---------------------------------------------------------------------------
// MI455X (gfx1250) fused multi-head attention forward.
//   D_MODEL=1024, H=16, D_K=64, B=2, T=2048.
// All GEMMs run on v_wmma_f32_16x16x32_bf16 (wave32). GEMMs stage block
// tiles in LDS via double-buffered GLOBAL_LOAD_ASYNC_TO_LDS_B128 (ASYNCcnt).
// Attention is flash-style streaming softmax (no materialized score tensor).
// Working set (<130MB) lives in the 192MB L2; kernels are matrix-pipe bound.
// ---------------------------------------------------------------------------

#define D_MODEL 1024
#define HEADS   16
#define DK      64
#define BATCH   2
#define SEQ     2048
#define BT      (BATCH * SEQ)   // 4096 rows of activations
#define KSTEP   32
#define LDA     40              // padded LDS row stride (ushorts): 80B, 16B-aligned

typedef unsigned short ushort_t;
typedef __attribute__((ext_vector_type(16))) __bf16 v16bf;
typedef __attribute__((ext_vector_type(8)))  float  v8f;
typedef __attribute__((ext_vector_type(4)))  int    v4i;

#if defined(__gfx1250__) && __has_builtin(__builtin_amdgcn_global_load_async_to_lds_b128)
#define USE_ASYNC 1
#else
#define USE_ASYNC 0
#endif

union Frag {
    v16bf v;
    uint4 u[2];
    ushort_t s[16];
};

__device__ __forceinline__ ushort_t f32_to_bf16(float f) {
    union { float f; unsigned u; } x; x.f = f;
    unsigned r = x.u + 0x7FFFu + ((x.u >> 16) & 1u);   // round-to-nearest-even
    return (ushort_t)(r >> 16);
}

// ---------------------------------------------------------------------------
// fp32 -> bf16 conversion pass
// ---------------------------------------------------------------------------
__global__ void cvt_bf16_kernel(const float* __restrict__ src,
                                ushort_t* __restrict__ dst, int n) {
    int i = blockIdx.x * blockDim.x + threadIdx.x;
    if (i < n) dst[i] = f32_to_bf16(src[i]);
}

// ---------------------------------------------------------------------------
// Generic bf16 GEMM:  C[m,n] = sum_k A[m,k] * W[n,k]   (i.e. y = x @ W^T)
// A: [M x K] bf16 row-major, W: [N x K] bf16 row-major.
// 256 threads = 8 waves (2x4), wave tile 32x32, block tile 64x128, Kstep 32.
// Block tiles are staged in LDS with double-buffered async copies: each
// thread issues 3 GLOBAL_LOAD_ASYNC_TO_LDS_B128 per K-step (1 A + 2 B
// chunks), pipelined one K-step ahead, synchronized with s_wait_asynccnt.
// Epilogue modes:
//   0: bf16, head-split  dst[((b*H+h)*T + t)*DK + d]      (q, k)
//   1: bf16, head-split transposed dst[((b*H+h)*DK + d)*T + t]  (v^T)
//   2: fp32 row-major dst[m*N + n]                         (final output)
// ---------------------------------------------------------------------------
__global__ __launch_bounds__(256)
void gemm_bf16_kernel(const ushort_t* __restrict__ A,
                      const ushort_t* __restrict__ W,
                      void* __restrict__ D,
                      int M, int N, int K, int mode)
{
    __shared__ __align__(16) ushort_t Abuf[2][64 * LDA];    // 2 x 5KB
    __shared__ __align__(16) ushort_t Bbuf[2][128 * LDA];   // 2 x 10KB

    const int tid  = threadIdx.x;
    const int lane = tid & 31;
    const int wave = tid >> 5;
    const int wm   = wave >> 2;               // 0..1
    const int wn   = wave & 3;                // 0..3
    const int l15  = lane & 15;
    const int grp  = lane >> 4;
    const int kbA  = grp * 8;                 // A-fragment chunk base
    const int kbB  = grp * 16;                // contiguous base (unused; B fragments load at kbA)
    const int mblk0 = blockIdx.y * 64;
    const int nblk0 = blockIdx.x * 128;

    // cooperative copy assignment: thread -> (row, 8-ushort column chunk)
    const int crow = tid >> 2;                // 0..63
    const int cch  = (tid & 3) * 8;           // 0,8,16,24
    const ushort_t* gA  = A + (size_t)(mblk0 + crow) * K + cch;
    const ushort_t* gB0 = W + (size_t)(nblk0 + crow) * K + cch;
    const ushort_t* gB1 = W + (size_t)(nblk0 + 64 + crow) * K + cch;
    const int lA  = crow * LDA + cch;
    const int lB0 = crow * LDA + cch;
    const int lB1 = (64 + crow) * LDA + cch;

    auto issue_tile = [&](int buf, int k0) {
#if USE_ASYNC
        __builtin_amdgcn_global_load_async_to_lds_b128(
            (v4i*)(gA  + k0), (v4i*)&Abuf[buf][lA],  0, 0);
        __builtin_amdgcn_global_load_async_to_lds_b128(
            (v4i*)(gB0 + k0), (v4i*)&Bbuf[buf][lB0], 0, 0);
        __builtin_amdgcn_global_load_async_to_lds_b128(
            (v4i*)(gB1 + k0), (v4i*)&Bbuf[buf][lB1], 0, 0);
#else
        *(uint4*)&Abuf[buf][lA]  = *(const uint4*)(gA  + k0);
        *(uint4*)&Bbuf[buf][lB0] = *(const uint4*)(gB0 + k0);
        *(uint4*)&Bbuf[buf][lB1] = *(const uint4*)(gB1 + k0);
#endif
    };

    v8f acc[2][2];
#pragma unroll
    for (int mt = 0; mt < 2; ++mt)
#pragma unroll
        for (int nt = 0; nt < 2; ++nt)
#pragma unroll
            for (int i = 0; i < 8; ++i) acc[mt][nt][i] = 0.0f;

    const int nk = K / KSTEP;
    issue_tile(0, 0);

    for (int kt = 0; kt < nk; ++kt) {
        const int cur = kt & 1;
        if (kt + 1 < nk) {
            issue_tile(cur ^ 1, (kt + 1) * KSTEP);
            // warm L2 for the tile after next (global_prefetch_b8)
            if (kt + 2 < nk) {
                __builtin_prefetch((const void*)(gA  + (kt + 2) * KSTEP), 0, 1);
                __builtin_prefetch((const void*)(gB0 + (kt + 2) * KSTEP), 0, 1);
            }
#if USE_ASYNC
            // 3 ops of tile kt+1 still in flight; tile kt fully landed
            asm volatile("s_wait_asynccnt 0x3" ::: "memory");
#endif
        } else {
#if USE_ASYNC
            asm volatile("s_wait_asynccnt 0x0" ::: "memory");
#endif
        }
        __syncthreads();

        Frag a[2], b[2];
#pragma unroll
        for (int mt = 0; mt < 2; ++mt) {
            const ushort_t* ap = &Abuf[cur][(wm * 32 + mt * 16 + l15) * LDA + kbA];
            a[mt].u[0] = *(const uint4*)(ap);
            a[mt].u[1] = *(const uint4*)(ap + 16);
        }
#pragma unroll
        for (int nt = 0; nt < 2; ++nt) {
            const ushort_t* bp = &Bbuf[cur][(wn * 32 + nt * 16 + l15) * LDA + kbA];
            b[nt].u[0] = *(const uint4*)(bp);
            b[nt].u[1] = *(const uint4*)(bp + 16);
        }
#pragma unroll
        for (int mt = 0; mt < 2; ++mt)
#pragma unroll
            for (int nt = 0; nt < 2; ++nt)
                acc[mt][nt] = __builtin_amdgcn_wmma_f32_16x16x32_bf16(
                    false, a[mt].v, false, b[nt].v,
                    (short)0, acc[mt][nt], false, false);
        __syncthreads();
    }

    // epilogue: C/D layout — VGPR r holds row (r) for lanes 0-15, (8+r) for 16-31
#pragma unroll
    for (int mt = 0; mt < 2; ++mt) {
#pragma unroll
        for (int nt = 0; nt < 2; ++nt) {
            const int ncol = nblk0 + wn * 32 + nt * 16 + l15;
#pragma unroll
            for (int r = 0; r < 8; ++r) {
                const int mrow = mblk0 + wm * 32 + mt * 16 + r + grp * 8;
                const float val = acc[mt][nt][r];
                if (mode == 2) {
                    ((float*)D)[(size_t)mrow * N + ncol] = val;
                } else {
                    const int bb = mrow >> 11;           // mrow / SEQ
                    const int tt = mrow & (SEQ - 1);
                    const int hh = ncol >> 6;            // ncol / DK
                    const int dd = ncol & (DK - 1);
                    ushort_t* du = (ushort_t*)D;
                    if (mode == 0)
                        du[(((size_t)bb * HEADS + hh) * SEQ + tt) * DK + dd] = f32_to_bf16(val);
                    else // mode 1: v transposed [B,H,DK,T]
                        du[(((size_t)bb * HEADS + hh) * DK + dd) * SEQ + tt] = f32_to_bf16(val);
                }
            }
        }
    }
}

// ---------------------------------------------------------------------------
// Flash attention core. One wave (32 threads) per 16-row query tile of one
// (batch, head). Streams 32-key tiles: scores via 4 WMMAs, online softmax
// (shfl_xor row reductions inside 16-lane halves), C->A transpose of the
// probability tile via 1KB LDS, then 4 WMMAs of P@V into the O accumulator.
// ---------------------------------------------------------------------------
__global__ __launch_bounds__(32)
void attn_kernel(const ushort_t* __restrict__ qh,   // [B*H, T, DK] bf16
                 const ushort_t* __restrict__ kh,   // [B*H, T, DK] bf16
                 const ushort_t* __restrict__ vT,   // [B*H, DK, T] bf16
                 ushort_t* __restrict__ ctx)        // [B*T, D_MODEL] bf16
{
    __shared__ __align__(16) ushort_t lds[16 * 32];

    const int lane = threadIdx.x & 31;
    const int l15  = lane & 15;
    const int grp  = lane >> 4;
    const int kbA  = grp * 8;
    const int kbB  = grp * 16;

    const int bh    = blockIdx.y;        // 0..B*H-1
    const int bb    = bh >> 4;
    const int hh    = bh & 15;
    const int qbase = blockIdx.x * 16;

    const ushort_t* qp = qh + (size_t)bh * SEQ * DK;
    const ushort_t* kp = kh + (size_t)bh * SEQ * DK;
    const ushort_t* vp = vT + (size_t)bh * DK * SEQ;

    // Q fragments for both K-steps (d = 0..31, 32..63), loaded once.
    Frag qf[2];
#pragma unroll
    for (int s = 0; s < 2; ++s) {
        const ushort_t* p = qp + (size_t)(qbase + l15) * DK + s * 32 + kbA;
        qf[s].u[0] = *(const uint4*)(p);
        qf[s].u[1] = *(const uint4*)(p + 16);
    }

    v8f o[4];
#pragma unroll
    for (int nt = 0; nt < 4; ++nt)
#pragma unroll
        for (int i = 0; i < 8; ++i) o[nt][i] = 0.0f;

    float mrow[8], lrow[8];
#pragma unroll
    for (int r = 0; r < 8; ++r) { mrow[r] = -1e30f; lrow[r] = 0.0f; }

    const float sc = 0.125f;   // 1/sqrt(DK)

    for (int j = 0; j <= qbase + 15; j += 32) {
        // ---- scores: S(16 x 32 keys) = q(16x64) @ k^T ----
        v8f s0, s1;
#pragma unroll
        for (int i = 0; i < 8; ++i) { s0[i] = 0.0f; s1[i] = 0.0f; }
        Frag kf;
#pragma unroll
        for (int s = 0; s < 2; ++s) {
            const ushort_t* p = kp + (size_t)(j + l15) * DK + s * 32 + kbA;
            kf.u[0] = *(const uint4*)(p);
            kf.u[1] = *(const uint4*)(p + 16);
            s0 = __builtin_amdgcn_wmma_f32_16x16x32_bf16(false, qf[s].v, false, kf.v,
                                                         (short)0, s0, false, false);
        }
#pragma unroll
        for (int s = 0; s < 2; ++s) {
            const ushort_t* p = kp + (size_t)(j + 16 + l15) * DK + s * 32 + kbA;
            kf.u[0] = *(const uint4*)(p);
            kf.u[1] = *(const uint4*)(p + 16);
            s1 = __builtin_amdgcn_wmma_f32_16x16x32_bf16(false, qf[s].v, false, kf.v,
                                                         (short)0, s1, false, false);
        }

        // ---- causal mask + online softmax ----
        float pr0[8], pr1[8];
        const int key0 = j + l15, key1 = key0 + 16;
#pragma unroll
        for (int r = 0; r < 8; ++r) {
            const int qrow = qbase + r + grp * 8;
            float x0 = s0[r] * sc; if (key0 > qrow) x0 = -1e30f;
            float x1 = s1[r] * sc; if (key1 > qrow) x1 = -1e30f;
            float mx = fmaxf(x0, x1);
#pragma unroll
            for (int off = 1; off < 16; off <<= 1)
                mx = fmaxf(mx, __shfl_xor(mx, off, 32));   // stays in 16-lane half
            const float mnew  = fmaxf(mrow[r], mx);
            const float alpha = __expf(mrow[r] - mnew);
            mrow[r] = mnew;
            const float p0 = __expf(x0 - mnew);
            const float p1 = __expf(x1 - mnew);
            float rs = p0 + p1;
#pragma unroll
            for (int off = 1; off < 16; off <<= 1)
                rs += __shfl_xor(rs, off, 32);
            lrow[r] = lrow[r] * alpha + rs;
#pragma unroll
            for (int nt = 0; nt < 4; ++nt) o[nt][r] *= alpha;
            pr0[r] = p0; pr1[r] = p1;
        }

        // ---- transpose P from C-layout to A-layout through LDS ----
#pragma unroll
        for (int r = 0; r < 8; ++r) {
            const int Mr = r + grp * 8;
            lds[Mr * 32 + l15]      = f32_to_bf16(pr0[r]);
            lds[Mr * 32 + 16 + l15] = f32_to_bf16(pr1[r]);
        }
        __syncthreads();   // single-wave WG: just orders DS ops
        Frag pf;
        {
            const ushort_t* p = &lds[l15 * 32 + kbA];
            pf.u[0] = *(const uint4*)(p);
            pf.u[1] = *(const uint4*)(p + 16);
        }
        __syncthreads();

        // ---- O += P @ V  (V stored transposed: contiguous along keys) ----
#pragma unroll
        for (int nt = 0; nt < 4; ++nt) {
            Frag vf;
            const ushort_t* p = vp + (size_t)(nt * 16 + l15) * SEQ + j + kbA;
            vf.u[0] = *(const uint4*)(p);
            vf.u[1] = *(const uint4*)(p + 16);
            o[nt] = __builtin_amdgcn_wmma_f32_16x16x32_bf16(false, pf.v, false, vf.v,
                                                            (short)0, o[nt], false, false);
        }
    }

    // ---- normalize and write context (bf16, merged-head layout) ----
#pragma unroll
    for (int nt = 0; nt < 4; ++nt) {
#pragma unroll
        for (int r = 0; r < 8; ++r) {
            const int Mr   = r + grp * 8;
            const int trow = qbase + Mr;
            const float val = o[nt][r] / lrow[r];
            ctx[((size_t)bb * SEQ + trow) * D_MODEL + hh * DK + nt * 16 + l15] =
                f32_to_bf16(val);
        }
    }
}

// ---------------------------------------------------------------------------
// Host-side launch. Inputs: Q,K,V,Wq,Wk,Wv,Wo,mask (mask handled analytically).
// Workspace layout (bf16 / ushort elements), ~64MB total.
// ---------------------------------------------------------------------------
extern "C" void kernel_launch(void* const* d_in, const int* in_sizes, int n_in,
                              void* d_out, int out_size, void* d_ws, size_t ws_size,
                              hipStream_t stream) {
    const float* Q  = (const float*)d_in[0];
    const float* K  = (const float*)d_in[1];
    const float* V  = (const float*)d_in[2];
    const float* Wq = (const float*)d_in[3];
    const float* Wk = (const float*)d_in[4];
    const float* Wv = (const float*)d_in[5];
    const float* Wo = (const float*)d_in[6];
    float* out = (float*)d_out;

    const size_t SZ_ACT = (size_t)BT * D_MODEL;       // 4M elements
    const size_t SZ_W   = (size_t)D_MODEL * D_MODEL;  // 1M elements

    ushort_t* ws  = (ushort_t*)d_ws;
    ushort_t* Qb  = ws;
    ushort_t* Kb  = Qb  + SZ_ACT;
    ushort_t* Vb  = Kb  + SZ_ACT;
    ushort_t* Wqb = Vb  + SZ_ACT;
    ushort_t* Wkb = Wqb + SZ_W;
    ushort_t* Wvb = Wkb + SZ_W;
    ushort_t* Wob = Wvb + SZ_W;
    ushort_t* qhd = Wob + SZ_W;
    ushort_t* khd = qhd + SZ_ACT;
    ushort_t* vTd = khd + SZ_ACT;
    ushort_t* ctx = vTd + SZ_ACT;

    const int TPB = 256;
    const int GA  = (int)((SZ_ACT + TPB - 1) / TPB);
    const int GW  = (int)((SZ_W   + TPB - 1) / TPB);
    cvt_bf16_kernel<<<GA, TPB, 0, stream>>>(Q,  Qb,  (int)SZ_ACT);
    cvt_bf16_kernel<<<GA, TPB, 0, stream>>>(K,  Kb,  (int)SZ_ACT);
    cvt_bf16_kernel<<<GA, TPB, 0, stream>>>(V,  Vb,  (int)SZ_ACT);
    cvt_bf16_kernel<<<GW, TPB, 0, stream>>>(Wq, Wqb, (int)SZ_W);
    cvt_bf16_kernel<<<GW, TPB, 0, stream>>>(Wk, Wkb, (int)SZ_W);
    cvt_bf16_kernel<<<GW, TPB, 0, stream>>>(Wv, Wvb, (int)SZ_W);
    cvt_bf16_kernel<<<GW, TPB, 0, stream>>>(Wo, Wob, (int)SZ_W);

    dim3 gg(D_MODEL / 128, BT / 64);   // (8, 64) blocks of 256 threads
    gemm_bf16_kernel<<<gg, 256, 0, stream>>>(Qb, Wqb, qhd, BT, D_MODEL, D_MODEL, 0);
    gemm_bf16_kernel<<<gg, 256, 0, stream>>>(Kb, Wkb, khd, BT, D_MODEL, D_MODEL, 0);
    gemm_bf16_kernel<<<gg, 256, 0, stream>>>(Vb, Wvb, vTd, BT, D_MODEL, D_MODEL, 1);

    attn_kernel<<<dim3(SEQ / 16, BATCH * HEADS), 32, 0, stream>>>(qhd, khd, vTd, ctx);

    gemm_bf16_kernel<<<gg, 256, 0, stream>>>(ctx, Wob, out, BT, D_MODEL, D_MODEL, 2);
}
